// SelfAttention_3521873183071
// MI455X (gfx1250) — matrix operand order repaired and checked by reading, not compiled, not run on hardware
//
#include <hip/hip_runtime.h>

// ---------------------------------------------------------------------------
// SAGAN self-attention for MI455X (gfx1250), wave32 + v_wmma_f32_16x16x32_bf16
//   B=4, C=256, Cqk=32, N=64*64=4096
//
// ws layout (bf16):  qT [B][N][32] | kT [B][N][32] | v [B][C][N]
// attn kernel: 1 workgroup (8 wave32) per (batch, 16-query tile)
//   - kT panels streamed by the Tensor Data Mover into a double-buffered LDS
//     ring (tensor_load_to_lds + s_wait_tensorcnt), overlapped with WMMA
//   - energy 16x4096 slab in LDS (CDNA5's 320KB/WGP LDS makes this fit)
//   - exact softmax in LDS, 1/sum + gamma folded into epilogue
// ---------------------------------------------------------------------------

typedef __attribute__((ext_vector_type(16))) __bf16 v16bf;
typedef __attribute__((ext_vector_type(8)))  __bf16 v8bf;
typedef __attribute__((ext_vector_type(8)))  float  v8f;

typedef __attribute__((ext_vector_type(4))) unsigned int u32x4;
typedef __attribute__((ext_vector_type(8))) int          i32x8;
typedef __attribute__((ext_vector_type(4))) int          i32x4;

#define NTOK 4096
#define CCH  256
#define CQK  32
#define PITCH 4104            // 4096 + 8 floats pad for the energy slab

// TDM staging ring for kT panels: 128 rows x 32 bf16, padded to 80B rows
#define KROWS      128                         // m-rows per TDM panel
#define KROW_BYTES 80                          // 64B data + 16B TDM pad
#define KBUF_BYTES (KROWS * KROW_BYTES)        // 10240 B per buffer
#define STAGE_OFF  ((16 * PITCH + 16) * 4)     // bytes: after slab + inv[16]

// ---- issue one TDM 2D tile load: kT panel [128 x 32] bf16 -> LDS ring ------
__device__ __forceinline__ void tdm_load_kpanel(const __bf16* gsrc,
                                                unsigned ldsByteOff) {
    unsigned long long ga = (unsigned long long)(size_t)gsrc;
    // D# group0: count=1 | lds_addr | global_addr[56:0] | type=2 ("image")
    u32x4 g0 = { 1u, ldsByteOff,
                 (unsigned)ga, (unsigned)(ga >> 32) | (2u << 30) };
    // D# group1:
    //  w0: data_size=1(2B)<<16 | pad_enable<<20 | pad_interval=3(16dw)<<22
    //      | pad_amount=3(4dw)<<25                       = 0x06D10000
    //  w1: tensor_dim0(=32) in [31:16]
    //  w2: tensor_dim1 low16 (=4096) in [31:16]
    //  w3: tile_dim0(=32) in [31:16]
    //  w4: tile_dim1(=128) in [15:0]
    //  w5: tensor_dim0_stride(=32)
    i32x8 g1 = { 0x06D10000, 32 << 16, 4096 << 16, 32 << 16, KROWS, 32, 0, 0 };
    i32x4 z4 = { 0, 0, 0, 0 };
#if __clang_major__ >= 23
    i32x8 z8 = { 0, 0, 0, 0, 0, 0, 0, 0 };
    __builtin_amdgcn_tensor_load_to_lds(g0, g1, z4, z4, z8, 0);
#else
    __builtin_amdgcn_tensor_load_to_lds(g0, g1, z4, z4, 0);
#endif
}

// ======================= projection: out = W @ x + b ========================
// D-tile [16 co x 16 n] = sum_{c-chunks of 32} A(W 16x32) * B(x 32x16)
// TOUT=1 -> out[n][co] (qT/kT: row-contiguous K=32 fragments later)
// TOUT=0 -> out[co][n] (v: row-contiguous K=m fragments later)
template <bool TOUT>
__global__ void __launch_bounds__(256)
proj_kernel(const float* __restrict__ x, const float* __restrict__ W,
            const float* __restrict__ bias, __bf16* __restrict__ out, int Co)
{
    const int tid  = threadIdx.x;
    const int lane = tid & 31;
    const int wave = tid >> 5;
    const int lo   = lane & 15;
    const int hi   = lane >> 4;

    const int tilesCo = Co >> 4;
    const int perB    = tilesCo * 256;          // 256 n-tiles
    const int t       = blockIdx.x * 8 + wave;
    const int b       = t / perB;
    const int rem     = t % perB;
    const int co0     = (rem >> 8) * 16;
    const int nb      = (rem & 255) * 16;

    const float* xb = x + (size_t)b * CCH * NTOK;

    v8f acc = {};
    for (int c0 = 0; c0 < CCH; c0 += 32) {
        // A = W tile: lane row co0+lo, K = c0 + hi*16 + i
        v16bf aw;
        const float* wp = W + (size_t)(co0 + lo) * CCH + c0 + hi * 16;
        #pragma unroll
        for (int i = 0; i < 8; ++i) aw[i]     = (__bf16)wp[i];
        #pragma unroll
        for (int i = 0; i < 8; ++i) aw[i + 8] = (__bf16)wp[8 + i];
        // B = x tile: lane col nb+lo, K = c0 + hi*16 + i   (coalesced per i)
        v16bf bx;
        const float* xp = xb + (size_t)(c0 + hi * 16) * NTOK + nb + lo;
        #pragma unroll
        for (int i = 0; i < 16; ++i) bx[i] = (__bf16)xp[(size_t)i * NTOK];

        acc = __builtin_amdgcn_wmma_f32_16x16x32_bf16(
                  false, aw, false, bx, (short)0, acc, false, false);
    }

    if (TOUT) {
        v8bf o;
        #pragma unroll
        for (int r = 0; r < 8; ++r)
            o[r] = (__bf16)(acc[r] + bias[co0 + hi * 8 + r]);
        *(v8bf*)(out + (size_t)(nb + lo) * Co + co0 + hi * 8) = o;  // 16B store
    } else {
        #pragma unroll
        for (int r = 0; r < 8; ++r) {
            const int co = co0 + hi * 8 + r;
            out[(size_t)co * NTOK + nb + lo] = (__bf16)(acc[r] + bias[co]);
        }
    }
}

// ============================ fused attention ===============================
extern "C" __global__ void __launch_bounds__(256)
attn_kernel(const __bf16* __restrict__ qh, const __bf16* __restrict__ kh,
            const __bf16* __restrict__ vh, const float* __restrict__ x,
            const float* __restrict__ gamma_p, float* __restrict__ out)
{
    extern __shared__ float lds[];   // [16][PITCH] slab | inv[16] | TDM ring
    float* inv = lds + 16 * PITCH;

    const int tid  = threadIdx.x;
    const int lane = tid & 31;
    const int wave = tid >> 5;
    const int lo   = lane & 15;
    const int hi   = lane >> 4;

    const int b  = blockIdx.x >> 8;
    const int n0 = (blockIdx.x & 255) * 16;

    const __bf16* qb = qh + (size_t)b * NTOK * CQK;
    const __bf16* kb = kh + (size_t)b * NTOK * CQK;
    const __bf16* vb = vh + (size_t)b * CCH * NTOK;
    const float*  xb = x  + (size_t)b * CCH * NTOK;
    float*        ob = out + (size_t)b * CCH * NTOK;

    // ---- A fragment: q tile (constant for the whole block) ----
    v16bf aq;
    {
        const __bf16* p = qb + (size_t)(n0 + lo) * CQK + hi * 16;
        v8bf a0 = *(const v8bf*)(p);
        v8bf a1 = *(const v8bf*)(p + 8);
        #pragma unroll
        for (int i = 0; i < 8; ++i) { aq[i] = a0[i]; aq[i + 8] = a1[i]; }
    }

    // ---- energy: E[16 x 4096] via TDM-staged kT panels -------------------
    // 32 iterations x 128 m-rows; wave w consumes rows [16w, 16w+16) of panel
    if (wave == 0)
        tdm_load_kpanel(kb, STAGE_OFF);                 // it=0 -> ring buf 0
    for (int it = 0; it < 32; ++it) {
        if (wave == 0)
            __builtin_amdgcn_s_wait_tensorcnt(0);       // panel it has landed
        __syncthreads();                                // publish to all waves
        if (wave == 0 && it + 1 < 32)
            tdm_load_kpanel(kb + (size_t)(it + 1) * KROWS * CQK,
                            STAGE_OFF + (unsigned)(((it + 1) & 1) * KBUF_BYTES));

        // B fragment from LDS ring: lane col m = 16*wave+lo, K=c = hi*16 + i
        const char* sp = (const char*)lds + STAGE_OFF + (it & 1) * KBUF_BYTES
                         + (wave * 16 + lo) * KROW_BYTES + hi * 32;
        v8bf b0 = *(const v8bf*)(sp);
        v8bf b1 = *(const v8bf*)(sp + 16);
        v16bf bk;
        #pragma unroll
        for (int i = 0; i < 8; ++i) { bk[i] = b0[i]; bk[i + 8] = b1[i]; }

        v8f e = {};
        e = __builtin_amdgcn_wmma_f32_16x16x32_bf16(
                false, aq, false, bk, (short)0, e, false, false);

        const int m0 = it * KROWS + wave * 16;
        #pragma unroll
        for (int r = 0; r < 8; ++r)
            lds[(size_t)(r + hi * 8) * PITCH + m0 + lo] = e[r];
        __syncthreads();             // reads of this ring buffer are done
    }

    // ---- exact softmax over each 4096-wide row; wave w owns rows 2w, 2w+1 --
    #pragma unroll
    for (int rr = 0; rr < 2; ++rr) {
        const int row = wave * 2 + rr;
        float* Pr = lds + (size_t)row * PITCH;
        float m = -3.0e38f;
        for (int j = lane; j < NTOK; j += 32) m = fmaxf(m, Pr[j]);
        #pragma unroll
        for (int off = 16; off > 0; off >>= 1) m = fmaxf(m, __shfl_xor(m, off, 32));
        float s = 0.f;
        for (int j = lane; j < NTOK; j += 32) {
            float e = __expf(Pr[j] - m);
            Pr[j] = e;
            s += e;
        }
        #pragma unroll
        for (int off = 16; off > 0; off >>= 1) s += __shfl_xor(s, off, 32);
        if (lane == 0) inv[row] = 1.0f / s;   // fold 1/sum into epilogue
    }
    __syncthreads();

    const float g = gamma_p[0];
    float isum[8];
    #pragma unroll
    for (int r = 0; r < 8; ++r) isum[r] = inv[hi * 8 + r];

    // ---- out tile [16 n x 32 c per wave] = P[16 x 4096] @ V^T ----
    const int c0 = wave * 32;                 // two adjacent c-tiles per wave
    v8f acc0 = {}, acc1 = {};
    for (int m0 = 0; m0 < NTOK; m0 += 32) {
        // A = P chunk from LDS (f32 -> bf16): lane row n0+lo, K = m0 + hi*16 + i
        v16bf ap;
        const float* Pr = lds + (size_t)lo * PITCH + m0 + hi * 16;
        #pragma unroll
        for (int i = 0; i < 8; ++i) ap[i]     = (__bf16)Pr[i];
        #pragma unroll
        for (int i = 0; i < 8; ++i) ap[i + 8] = (__bf16)Pr[8 + i];

        // B = V^T chunks: lane col c, K = m0 + hi*16 + i (contiguous in m)
        const __bf16* p0 = vb + (size_t)(c0 + lo) * NTOK + m0 + hi * 16;
        const __bf16* p1 = p0 + (size_t)16 * NTOK;
        __builtin_prefetch((const void*)(p0 + 256), 0, 1);
        v8bf b00 = *(const v8bf*)(p0);
        v8bf b01 = *(const v8bf*)(p0 + 8);
        v8bf b10 = *(const v8bf*)(p1);
        v8bf b11 = *(const v8bf*)(p1 + 8);
        v16bf bv0, bv1;
        #pragma unroll
        for (int i = 0; i < 8; ++i) {
            bv0[i] = b00[i]; bv0[i + 8] = b01[i];
            bv1[i] = b10[i]; bv1[i + 8] = b11[i];
        }
        acc0 = __builtin_amdgcn_wmma_f32_16x16x32_bf16(
                   false, ap, false, bv0, (short)0, acc0, false, false);
        acc1 = __builtin_amdgcn_wmma_f32_16x16x32_bf16(
                   false, ap, false, bv1, (short)0, acc1, false, false);
    }

    // ---- epilogue: out[c][n] = gamma * (acc * 1/sum) + x[c][n] ----
    #pragma unroll
    for (int half = 0; half < 2; ++half) {
        const int c = c0 + half * 16 + lo;
        float* op       = ob + (size_t)c * NTOK + n0 + hi * 8;
        const float* xp = xb + (size_t)c * NTOK + n0 + hi * 8;
        const v8f& a = half ? acc1 : acc0;
        #pragma unroll
        for (int r = 0; r < 8; ++r)
            op[r] = g * (a[r] * isum[r]) + xp[r];
    }
}

// =============================== launcher ===================================
extern "C" void kernel_launch(void* const* d_in, const int* in_sizes, int n_in,
                              void* d_out, int out_size, void* d_ws, size_t ws_size,
                              hipStream_t stream) {
    (void)in_sizes; (void)n_in; (void)out_size; (void)ws_size;
    const float* x     = (const float*)d_in[0];
    const float* Wq    = (const float*)d_in[1];
    const float* bq    = (const float*)d_in[2];
    const float* Wk    = (const float*)d_in[3];
    const float* bk    = (const float*)d_in[4];
    const float* Wv    = (const float*)d_in[5];
    const float* bv    = (const float*)d_in[6];
    const float* gamma = (const float*)d_in[7];
    float* out = (float*)d_out;

    const size_t QK_BYTES = (size_t)4 * NTOK * CQK * sizeof(__bf16);  // 1 MB
    __bf16* qh = (__bf16*)d_ws;
    __bf16* kh = (__bf16*)((char*)d_ws + QK_BYTES);
    __bf16* vh = (__bf16*)((char*)d_ws + 2 * QK_BYTES);               // 8 MB

    // projections: 8 wave-tiles per 256-thread block
    proj_kernel<true ><<<256,  256, 0, stream>>>(x, Wq, bq, qh, CQK);   // qT
    proj_kernel<true ><<<256,  256, 0, stream>>>(x, Wk, bk, kh, CQK);   // kT
    proj_kernel<false><<<2048, 256, 0, stream>>>(x, Wv, bv, vh, CCH);   // v

    // fused attention: slab (256.5KB) + inv + TDM ring (20KB) = 283200 B LDS
    const size_t ldsBytes = (size_t)STAGE_OFF + 2 * KBUF_BYTES;
    hipFuncSetAttribute((const void*)attn_kernel,
                        hipFuncAttributeMaxDynamicSharedMemorySize, (int)ldsBytes);
    attn_kernel<<<4 * 256, 256, ldsBytes, stream>>>(qh, kh, vh, x, gamma, out);
}
